// CharRNN_86792699118074
// MI455X (gfx1250) — matrix operand order repaired and checked by reading, not compiled, not run on hardware
//
#include <hip/hip_runtime.h>
#include <hip/hip_bf16.h>
#include <math.h>

// Problem constants (match reference).
#define B_  32
#define L_  1024
#define H_  1024
#define E_  512
#define V_  8192
#define EH_ (E_ + H_)   // 1536

typedef __bf16 bf16_t;
typedef __attribute__((ext_vector_type(16))) __bf16       v16bf;
typedef __attribute__((ext_vector_type(8)))  float        v8f;
typedef __attribute__((ext_vector_type(4)))  unsigned int v4u;

// 32-byte WMMA operand fragment viewed as two 16-byte loads.
union Frag32 {
    v16bf v;
    v4u   u[2];
};

// ---------------------------------------------------------------------------
// Kernel 1: convert embedding -> bf16, W_h -> bf16 transposed [H][E+H],
// W_o -> bf16 transposed [V][H], hidden0 -> bf16, zero the barrier counter.
// Transposed weights make each lane's B-fragment two contiguous 16B loads.
// ---------------------------------------------------------------------------
__global__ void __launch_bounds__(256)
prep_kernel(const float* __restrict__ emb,
            const float* __restrict__ Wh,
            const float* __restrict__ Wo,
            const float* __restrict__ h0,
            bf16_t* __restrict__ embB,
            bf16_t* __restrict__ WhT,
            bf16_t* __restrict__ WoT,
            bf16_t* __restrict__ hbuf,
            unsigned int* __restrict__ bar)
{
    const size_t tid    = (size_t)blockIdx.x * blockDim.x + threadIdx.x;
    const size_t stride = (size_t)gridDim.x * blockDim.x;

    for (size_t i = tid; i < (size_t)V_ * E_; i += stride)
        embB[i] = (bf16_t)emb[i];

    for (size_t i = tid; i < (size_t)H_ * EH_; i += stride) {
        const size_t n = i / EH_, k = i % EH_;
        WhT[i] = (bf16_t)Wh[k * H_ + n];          // WhT[n][k] = Wh[k][n]
    }

    for (size_t i = tid; i < (size_t)V_ * H_; i += stride) {
        const size_t v = i / H_, k = i % H_;
        WoT[i] = (bf16_t)Wo[k * V_ + v];          // WoT[v][k] = Wo[k][v]
    }

    for (size_t i = tid; i < (size_t)B_ * H_; i += stride)
        hbuf[i] = (bf16_t)h0[i];                  // initial hidden (zeros)

    if (tid == 0) *bar = 0u;
}

// ---------------------------------------------------------------------------
// Kernel 2: persistent recurrent scan.
// 128 waves total = 2 M-tiles x 64 N-tiles of the [32 x 1024] hidden state.
// Per step: h_new = tanh([embed(x_t), h] @ W_h + b_h), K = 1536 -> 48 WMMAs.
// Device-wide sync per step via a monotonic atomic counter (32 workgroups,
// all co-resident: 32 blocks x 128 threads).
// ---------------------------------------------------------------------------
__global__ void __launch_bounds__(128)
rnn_scan_kernel(const int*    __restrict__ x,      // [B, L]
                const float*  __restrict__ bh,     // [H]
                const bf16_t* __restrict__ embB,   // [V, E]
                const bf16_t* __restrict__ WhT,    // [H, E+H]
                bf16_t*       __restrict__ hbuf,   // 2 * [B, H] double buffer
                bf16_t*       __restrict__ outs,   // [B, L, H] bf16 hidden outs
                float*        __restrict__ final_h,// [B, H] fp32 (d_out tail)
                unsigned int* __restrict__ bar)
{
    const int lane  = threadIdx.x & 31;
    const int wave  = blockIdx.x * (blockDim.x >> 5) + (threadIdx.x >> 5); // 0..127
    const int mtile = wave >> 6;            // 0..1
    const int ntile = wave & 63;            // 0..63
    const int m0    = mtile * 16;
    const int n0    = ntile * 16;
    const int half  = lane >> 4;            // K-half select per ISA layout
    const int lan   = lane & 15;
    const int m     = m0 + lan;             // A-matrix row (batch) this lane feeds
    const int n     = n0 + lan;             // output column this lane owns
    const float bias = bh[n];
    const bf16_t* wrow = WhT + (size_t)n * EH_;   // B-fragment source (column n)

    for (int t = 0; t < L_; ++t) {
        const bf16_t* hcur  = hbuf + (size_t)(t & 1) * (B_ * H_);
        bf16_t*       hnext = hbuf + (size_t)((t + 1) & 1) * (B_ * H_);

        const int token = x[m * L_ + t];
        const bf16_t* arow_e = embB + (size_t)token * E_;  // embedded part, K<512
        const bf16_t* arow_h = hcur + (size_t)m * H_;      // hidden part,  K>=512

        v8f c = {};
        #pragma unroll 4
        for (int kk = 0; kk < EH_ / 32; ++kk) {
            const int k0 = kk * 32;
            // A fragment: 16-bit A 16x32 layout -> lane holds row m,
            // K = k0 + half*8 .. +7 and K = k0 + half*8 + 16 .. +7.
            const bf16_t* abase = (k0 < E_) ? (arow_e + k0) : (arow_h + (k0 - E_));
            const bf16_t* pa = abase + half * 8;
            Frag32 fa;
            fa.u[0] = *(const v4u*)pa;
            fa.u[1] = *(const v4u*)(pa + 16);
            // B fragment: lane holds column n, K = k0 + half*8 .. +7 and k0 + half*8 + 16 .. +7.
            const bf16_t* pb = wrow + k0 + half * 8;
            Frag32 fb;
            fb.u[0] = *(const v4u*)pb;
            fb.u[1] = *(const v4u*)(pb + 16);

            c = __builtin_amdgcn_wmma_f32_16x16x32_bf16(
                    false, fa.v, false, fb.v, (short)0, c, false, false);
        }

        // Bias + tanh; D layout: VGPR r -> row m0 + r + 8*half, col n.
        #pragma unroll
        for (int r = 0; r < 8; ++r) {
            const int mr = m0 + r + half * 8;
            const float hv = tanhf(c[r] + bias);
            const bf16_t hb = (bf16_t)hv;
            hnext[(size_t)mr * H_ + n] = hb;
            outs[(size_t)mr * (L_ * H_) + (size_t)t * H_ + n] = hb;
            if (t == L_ - 1) final_h[(size_t)mr * H_ + n] = hv;
        }

        // Device-wide step barrier (monotonic counter, 32 workgroups).
        if (t != L_ - 1) {
            __threadfence();
            __syncthreads();
            if (threadIdx.x == 0) {
                __hip_atomic_fetch_add(bar, 1u, __ATOMIC_RELEASE,
                                       __HIP_MEMORY_SCOPE_AGENT);
                const unsigned int target = 32u * (unsigned int)(t + 1);
                while (__hip_atomic_load(bar, __ATOMIC_ACQUIRE,
                                         __HIP_MEMORY_SCOPE_AGENT) < target)
                    __builtin_amdgcn_s_sleep(1);
            }
            __syncthreads();
            __threadfence();
        }
    }
}

// ---------------------------------------------------------------------------
// Kernel 3: logits = outs @ W_o + b_o.  [32768 x 1024] @ [1024 x 8192].
// Each 8-wave block computes a 128x128 output tile: the 8 waves SHARE the
// same 128 columns (B operand, staged in LDS) and take 8 distinct 16-row
// M-tiles. B is cooperatively loaded per 128-wide K-chunk (16 KB: K 0..31, 64..95), padded to
// a 136-element row stride so B-fragment ds_load_b128s are bank-conflict-free.
// ---------------------------------------------------------------------------
#define KC_  128            // K-chunk staged in LDS
#define KCP_ 136            // padded LDS row stride (272 B -> 4-bank lane step)
__global__ void __launch_bounds__(256)
logits_kernel(const bf16_t* __restrict__ outs,  // [B*L, H]
              const bf16_t* __restrict__ WoT,   // [V, H]
              const float*  __restrict__ bo,    // [V]
              float*        __restrict__ logits)// [B*L, V]
{
    __shared__ bf16_t Bs[128 * KCP_];           // 34 KB

    const int lane = threadIdx.x & 31;
    const int wv   = threadIdx.x >> 5;          // wave in block, 0..7
    const int nb   = blockIdx.x & 63;           // N-block: 64 x 128 cols
    const int mb   = blockIdx.x >> 6;           // M-block: 256 x 128 rows
    const int m0   = mb * 128 + wv * 16;        // this wave's 16 rows
    const int n0   = nb * 128;                  // block's 128 columns
    const int half = lane >> 4;
    const int lan  = lane & 15;

    // Cooperative B-copy assignment: thread -> (column, 64-wide k-half).
    const int ccol  = threadIdx.x >> 1;         // 0..127
    const int chalf = threadIdx.x & 1;          // 0..1
    const bf16_t* csrc_base = WoT + (size_t)(n0 + ccol) * H_ + chalf * 64;
    bf16_t*       cdst      = Bs + ccol * KCP_ + chalf * 64;

    const bf16_t* arow = outs + (size_t)(m0 + lan) * H_;

    v8f acc[8];
    #pragma unroll
    for (int j = 0; j < 8; ++j) { v8f z = {}; acc[j] = z; }

    for (int kc = 0; kc < H_ / KC_; ++kc) {
        const int kbase = kc * KC_;

        // Stage B chunk: 128 cols x 64 K of WoT (K 0..31, 64..95) -> LDS (64 B per thread).
        {
            const bf16_t* src = csrc_base + kbase;
            #pragma unroll
            for (int q = 0; q < 4; ++q)
                *(v4u*)(cdst + q * 8) = *(const v4u*)(src + q * 8);
            // Prefetch next chunk's source lines into cache.
            if (kc + 1 < H_ / KC_)
                __builtin_prefetch(src + KC_, 0, 1);
        }
        __syncthreads();

        #pragma unroll
        for (int kk = 0; kk < KC_ / 32; ++kk) {
            const int k0 = kbase + kk * 32;
            Frag32 fa;
            const bf16_t* pa = arow + k0 + half * 8;
            fa.u[0] = *(const v4u*)pa;
            fa.u[1] = *(const v4u*)(pa + 16);
            #pragma unroll
            for (int j = 0; j < 8; ++j) {
                // B fragment from LDS: column n0 + j*16 + lan, K = half*8 .. +7 and half*8 + 16 .. +7.
                const bf16_t* pb = Bs + (j * 16 + lan) * KCP_ + kk * 32 + half * 8;
                Frag32 fb;
                fb.u[0] = *(const v4u*)pb;
                fb.u[1] = *(const v4u*)(pb + 16);
                acc[j] = __builtin_amdgcn_wmma_f32_16x16x32_bf16(
                             false, fa.v, false, fb.v, (short)0, acc[j],
                             false, false);
            }
        }
        __syncthreads();
    }

    #pragma unroll
    for (int j = 0; j < 8; ++j) {
        const int n = n0 + j * 16 + lan;
        const float bias = bo[n];
        #pragma unroll
        for (int r = 0; r < 8; ++r) {
            const int mr = m0 + r + half * 8;
            logits[(size_t)mr * V_ + n] = acc[j][r] + bias;
        }
    }
}

// ---------------------------------------------------------------------------
// Launch: prep -> persistent scan -> logits, all stream-ordered.
// Workspace layout (bytes): needs >= ~92.2 MB.
// ---------------------------------------------------------------------------
extern "C" void kernel_launch(void* const* d_in, const int* in_sizes, int n_in,
                              void* d_out, int out_size, void* d_ws, size_t ws_size,
                              hipStream_t stream) {
    const int*   x      = (const int*)  d_in[0];  // [B, L]
    const float* hidden = (const float*)d_in[1];  // [B, H]
    const float* emb    = (const float*)d_in[2];  // [V, E]
    const float* Wh     = (const float*)d_in[3];  // [E+H, H]
    const float* bh     = (const float*)d_in[4];  // [H]
    const float* Wo     = (const float*)d_in[5];  // [H, V]
    const float* bo     = (const float*)d_in[6];  // [V]

    float* logits  = (float*)d_out;                         // [B, L, V]
    float* final_h = logits + (size_t)B_ * L_ * V_;         // [B, H]

    char* ws = (char*)d_ws;
    bf16_t*       embB = (bf16_t*)(ws);                          //  8 MB
    bf16_t*       WhT  = (bf16_t*)(ws + (size_t)8  * (1u << 20));//  3 MB
    bf16_t*       WoT  = (bf16_t*)(ws + (size_t)11 * (1u << 20));// 16 MB
    bf16_t*       outs = (bf16_t*)(ws + (size_t)27 * (1u << 20));// 64 MB
    bf16_t*       hbuf = (bf16_t*)(ws + (size_t)91 * (1u << 20));// 128 KB (x2)
    unsigned int* bar  = (unsigned int*)(ws + (size_t)92 * (1u << 20));

    prep_kernel<<<2048, 256, 0, stream>>>(emb, Wh, Wo, hidden,
                                          embB, WhT, WoT, hbuf, bar);

    rnn_scan_kernel<<<32, 128, 0, stream>>>(x, bh, embB, WhT,
                                            hbuf, outs, final_h, bar);

    // 256 M-blocks * 64 N-blocks of 128x128 output tiles.
    logits_kernel<<<16384, 256, 0, stream>>>(outs, WoT, bo, logits);
}
